// StandardAttention_53352083751000
// MI455X (gfx1250) — matrix operand order repaired and checked by reading, not compiled, not run on hardware
//
#include <hip/hip_runtime.h>
#include <hip/hip_bf16.h>

typedef __attribute__((ext_vector_type(16))) __bf16 v16bf;
typedef __attribute__((ext_vector_type(8)))  __bf16 v8bf;
typedef __attribute__((ext_vector_type(8)))  float  v8f;
typedef __attribute__((ext_vector_type(4)))  unsigned int v4u;
typedef __attribute__((ext_vector_type(8)))  int    v8i;
typedef __attribute__((ext_vector_type(4)))  int    v4i;

#define D_MODEL 1024
#define SEQ     2048
#define NB      2
#define NH      16
#define HD      64
#define MROWS   4096  // B*L

union V16U { v16bf v; v8bf h[2]; };

// A-fragment (16x32 bf16, MxK): lane m=lane&15 holds K in [8h,8h+8) and [16+8h,16+8h+8), h=lane>>4
__device__ __forceinline__ v16bf load_a_frag(const __bf16* base, int stride, int lane) {
  int m  = lane & 15;
  int h8 = (lane >> 4) * 8;
  V16U u;
  u.h[0] = *(const v8bf*)(base + m * stride + h8);
  u.h[1] = *(const v8bf*)(base + m * stride + 16 + h8);
  return u.v;
}

// B-fragment (32x16 bf16, KxN) from an N-major (transposed) tile: lane n=lane&15,
// K in [8h,8h+8) and [16+8h,16+8h+8), h=lane>>4
__device__ __forceinline__ v16bf load_b_frag(const __bf16* baseT, int stride, int lane) {
  int n = lane & 15;
  int k = (lane >> 4) * 8;
  V16U u;
  u.h[0] = *(const v8bf*)(baseT + n * stride + k);
  u.h[1] = *(const v8bf*)(baseT + n * stride + k + 16);
  return u.v;
}

__device__ __forceinline__ v8f wmma_bf16(v16bf a, v16bf b, v8f c) {
  return __builtin_amdgcn_wmma_f32_16x16x32_bf16(false, a, false, b, (short)0, c, false, false);
}

// ---- Tensor Data Mover: 2D tile (bf16 elements) global -> LDS, per ISA D# layout (§8) ----
// dims/strides in elements. LDS fill order: x fastest => LDS tile is [y][x], row stride = tile_d0.
__device__ __forceinline__ void tdm_load_2d(const __bf16* lds_dst, const __bf16* gsrc,
                                            unsigned tensor_d0, unsigned tensor_d1,
                                            unsigned tile_d0, unsigned tile_d1,
                                            unsigned long long stride0) {
  unsigned lds_off = (unsigned)(unsigned long long)(uintptr_t)lds_dst;  // low 32b = LDS byte addr
  unsigned long long ga = (unsigned long long)(uintptr_t)gsrc;
  v4u g0;
  g0[0] = 1u;                                               // count=1, user mode
  g0[1] = lds_off;                                          // lds_addr (bytes)
  g0[2] = (unsigned)(ga & 0xFFFFFFFFu);                     // global_addr[31:0]
  g0[3] = (unsigned)((ga >> 32) & 0x1FFFFFFu) | 0x80000000u; // global_addr[56:32] | type=2
  v8i g1;
  g1[0] = (int)(1u << 16);                                  // data_size=1 (2 bytes/elem)
  g1[1] = (int)((tensor_d0 & 0xFFFFu) << 16);               // tensor_dim0[15:0] @ bits 63:48
  g1[2] = (int)((tensor_d0 >> 16) | ((tensor_d1 & 0xFFFFu) << 16));
  g1[3] = (int)((tensor_d1 >> 16) | (tile_d0 << 16));       // tile_dim0 @ bits 127:112
  g1[4] = (int)tile_d1;                                     // tile_dim1 (tile_dim2=0)
  g1[5] = (int)(unsigned)(stride0 & 0xFFFFFFFFu);           // tensor_dim0_stride[31:0]
  g1[6] = (int)(unsigned)((stride0 >> 32) & 0xFFFFu);       // stride0[47:32], stride1=0
  g1[7] = 0;
  v4i z4 = {0, 0, 0, 0};
  v8i z8 = {0, 0, 0, 0, 0, 0, 0, 0};
  __builtin_amdgcn_tensor_load_to_lds(g0, g1, z4, z4, z8, 0);  // 6-arg toolchain form
}

// one 64x64-tile GEMM step: 2x2 WMMA subtiles per wave from LDS-staged A/B
__device__ __forceinline__ void gemm_step(const __bf16* __restrict__ Ab,
                                          const __bf16* __restrict__ Bb,
                                          int wrow, int wcol, int lane, v8f c[2][2]) {
  v16bf a0 = load_a_frag(Ab + (wrow +  0) * 32, 32, lane);
  v16bf a1 = load_a_frag(Ab + (wrow + 16) * 32, 32, lane);
  v16bf b0 = load_b_frag(Bb + (wcol +  0) * 32, 32, lane);
  v16bf b1 = load_b_frag(Bb + (wcol + 16) * 32, 32, lane);
  c[0][0] = wmma_bf16(a0, b0, c[0][0]);
  c[0][1] = wmma_bf16(a0, b1, c[0][1]);
  c[1][0] = wmma_bf16(a1, b0, c[1][0]);
  c[1][1] = wmma_bf16(a1, b1, c[1][1]);
}

// ---------------- fp32 -> bf16 conversion ----------------
__global__ void mha_cvt_kernel(const float* __restrict__ s, __bf16* __restrict__ d, int n) {
  int i = blockIdx.x * 256 + threadIdx.x;
  if (i < n) d[i] = (__bf16)s[i];
}

// ---------------- fp32 [K][N] -> bf16 transposed [N][K] (one-time, LDS-tiled) ----------------
__global__ __launch_bounds__(256) void mha_cvtT_kernel(const float* __restrict__ src,
                                                       __bf16* __restrict__ dst) {
  __shared__ __bf16 t[32][33];
  int bx = blockIdx.x * 32, by = blockIdx.y * 32;  // bx: N, by: K
  int tx = threadIdx.x, ty = threadIdx.y;          // (32, 8)
#pragma unroll
  for (int i = 0; i < 4; i++)
    t[ty + 8 * i][tx] = (__bf16)src[(size_t)(by + ty + 8 * i) * D_MODEL + bx + tx];
  __syncthreads();
#pragma unroll
  for (int i = 0; i < 4; i++)
    dst[(size_t)(bx + ty + 8 * i) * D_MODEL + by + tx] = t[tx][ty + 8 * i];
}

// ---------------- projection: C = X @ W  (W given transposed: Wt[N][K]) ----------------
// mode 0: store bf16 to [B,H,L,hd] with scale (Q, K)
// mode 1: store bf16 to [B,H,hd,L] (V, pre-transposed for attention)
__global__ __launch_bounds__(128) void mha_proj_qkv_kernel(const __bf16* __restrict__ X,
                                                           const __bf16* __restrict__ Wt,
                                                           __bf16* __restrict__ dst,
                                                           float scale, int mode) {
  __shared__ __align__(32) __bf16 At[2][64 * 32];   // [row][k]
  __shared__ __align__(32) __bf16 Bt[2][64 * 32];   // [n][k]
  int tid = threadIdx.x, lane = tid & 31, wave = tid >> 5;
  int row0 = blockIdx.y * 64, col0 = blockIdx.x * 64;
  const int NK = D_MODEL / 32;                      // 32 k-steps (even)
  const __bf16* Arow = X  + (size_t)row0 * D_MODEL;
  const __bf16* Brow = Wt + (size_t)col0 * D_MODEL;
  v8f c[2][2];
#pragma unroll
  for (int i = 0; i < 2; i++)
#pragma unroll
    for (int j = 0; j < 2; j++)
#pragma unroll
      for (int r = 0; r < 8; r++) c[i][j][r] = 0.f;

  int wrow = (wave >> 1) * 32, wcol = (wave & 1) * 32;

  if (wave == 0) {  // prologue: DMA tile 0 into buffer 0
    tdm_load_2d(At[0], Arow, D_MODEL, 64, 32, 64, D_MODEL);
    tdm_load_2d(Bt[0], Brow, D_MODEL, 64, 32, 64, D_MODEL);
  }

  // k-loop unrolled x2 so the ping-pong buffers are compile-time constants
  for (int ks = 0; ks < NK; ks += 2) {
    if (wave == 0) {  // prefetch ks+1 into buffer 1 (ks+1 <= NK-1 always)
      int kn = (ks + 1) * 32;
      tdm_load_2d(At[1], Arow + kn, D_MODEL, 64, 32, 64, D_MODEL);
      tdm_load_2d(Bt[1], Brow + kn, D_MODEL, 64, 32, 64, D_MODEL);
      __builtin_amdgcn_s_wait_tensorcnt(2);  // tile ks complete (in-order)
    }
    __syncthreads();
    gemm_step(At[0], Bt[0], wrow, wcol, lane, c);
    __syncthreads();

    if (wave == 0) {
      if (ks + 2 < NK) {  // prefetch ks+2 into buffer 0
        int kn = (ks + 2) * 32;
        tdm_load_2d(At[0], Arow + kn, D_MODEL, 64, 32, 64, D_MODEL);
        tdm_load_2d(Bt[0], Brow + kn, D_MODEL, 64, 32, 64, D_MODEL);
        __builtin_amdgcn_s_wait_tensorcnt(2);
      } else {
        __builtin_amdgcn_s_wait_tensorcnt(0);
      }
    }
    __syncthreads();
    gemm_step(At[1], Bt[1], wrow, wcol, lane, c);
    __syncthreads();
  }

#pragma unroll
  for (int mi = 0; mi < 2; mi++)
#pragma unroll
    for (int ni = 0; ni < 2; ni++)
#pragma unroll
      for (int r = 0; r < 8; r++) {
        int row = row0 + wrow + mi * 16 + r + 8 * (lane >> 4);
        int col = col0 + wcol + ni * 16 + (lane & 15);
        int b = row >> 11, l = row & (SEQ - 1);
        int h = col >> 6,  d = col & (HD - 1);
        float v = c[mi][ni][r] * scale;
        if (mode == 0)
          dst[((size_t)((b * NH + h) * SEQ + l)) * HD + d] = (__bf16)v;
        else
          dst[((size_t)((b * NH + h) * HD + d)) * SEQ + l] = (__bf16)v;
      }
}

// ---------------- Flash attention: per block = one 64-row Q tile of one (b,h) ----------------
__global__ __launch_bounds__(128) void mha_attn_kernel(const __bf16* __restrict__ Q,
                                                       const __bf16* __restrict__ K,
                                                       const __bf16* __restrict__ VT,
                                                       __bf16* __restrict__ ctx) {
  __shared__ __align__(32) __bf16 Kt[2][64 * 64];   // [kv][d]
  __shared__ __align__(32) __bf16 Vt[2][64 * 64];   // [d][kv] (source pre-transposed)
  __shared__ __align__(32) __bf16 Pt[4][16 * 64];   // per-wave P strip, A-frag layout
  int tid = threadIdx.x, lane = tid & 31, wave = tid >> 5;
  int bh = blockIdx.y, qt = blockIdx.x;
  const __bf16* qp  = Q  + (size_t)bh * SEQ * HD;
  const __bf16* kp  = K  + (size_t)bh * SEQ * HD;
  const __bf16* vtp = VT + (size_t)bh * HD * SEQ;
  int qbase = qt * 64 + wave * 16;

  v16bf aq0 = load_a_frag(qp + (size_t)qbase * HD + 0,  HD, lane);
  v16bf aq1 = load_a_frag(qp + (size_t)qbase * HD + 32, HD, lane);

  float mrun[8], lrun[8];
  v8f acc[4];
#pragma unroll
  for (int r = 0; r < 8; r++) { mrun[r] = -1e30f; lrun[r] = 0.f; }
#pragma unroll
  for (int t = 0; t < 4; t++)
#pragma unroll
    for (int r = 0; r < 8; r++) acc[t][r] = 0.f;

  if (wave == 0) {  // prologue: DMA kv-tile 0
    tdm_load_2d(Kt[0], kp,  HD,  SEQ, 64, 64, HD);
    tdm_load_2d(Vt[0], vtp, SEQ, HD,  64, 64, SEQ);
  }

  for (int kvt = 0; kvt <= qt; ++kvt) {
    int kvbase = kvt * 64;
    if (wave == 0) {
      if (kvt < qt) {  // prefetch next kv tile
        int nb = kvbase + 64;
        tdm_load_2d(Kt[(kvt + 1) & 1], kp + (size_t)nb * HD, HD,  SEQ, 64, 64, HD);
        tdm_load_2d(Vt[(kvt + 1) & 1], vtp + nb,             SEQ, HD,  64, 64, SEQ);
        __builtin_amdgcn_s_wait_tensorcnt(2);
      } else {
        __builtin_amdgcn_s_wait_tensorcnt(0);
      }
    }
    __syncthreads();
    const __bf16* Kb = Kt[kvt & 1];
    const __bf16* Vb = Vt[kvt & 1];

    // S = Q * K^T  (16 x 64 per wave, 4 N-tiles, hd=64 => two k-steps)
    v8f s[4];
#pragma unroll
    for (int nt = 0; nt < 4; nt++) {
      v8f z;
#pragma unroll
      for (int r = 0; r < 8; r++) z[r] = 0.f;
      v16bf bk0 = load_b_frag(Kb + nt * 16 * 64 + 0,  64, lane);
      v16bf bk1 = load_b_frag(Kb + nt * 16 * 64 + 32, 64, lane);
      z = wmma_bf16(aq0, bk0, z);
      s[nt] = wmma_bf16(aq1, bk1, z);
    }

    if (kvt == qt) {  // causal mask on the diagonal tile
#pragma unroll
      for (int nt = 0; nt < 4; nt++)
#pragma unroll
        for (int r = 0; r < 8; r++) {
          int kvg = kvbase + nt * 16 + (lane & 15);
          int qg  = qbase + r + 8 * (lane >> 4);
          if (kvg > qg) s[nt][r] = -1e30f;
        }
    }

    // online softmax: per-row stats via cross-lane xor reductions within 16-lane halves
#pragma unroll
    for (int r = 0; r < 8; r++) {
      float mx = fmaxf(fmaxf(s[0][r], s[1][r]), fmaxf(s[2][r], s[3][r]));
#pragma unroll
      for (int mk = 1; mk < 16; mk <<= 1) mx = fmaxf(mx, __shfl_xor(mx, mk, 32));
      float mnew = fmaxf(mrun[r], mx);
      float corr = __expf(mrun[r] - mnew);
      lrun[r] *= corr;
#pragma unroll
      for (int t = 0; t < 4; t++) acc[t][r] *= corr;
      float rs = 0.f;
#pragma unroll
      for (int nt = 0; nt < 4; nt++) {
        float p = __expf(s[nt][r] - mnew);
        s[nt][r] = p;
        rs += p;
      }
#pragma unroll
      for (int mk = 1; mk < 16; mk <<= 1) rs += __shfl_xor(rs, mk, 32);
      lrun[r] += rs;
      mrun[r] = mnew;
    }

    // lay P into LDS row-major (A-frag layout), per-wave region -> no block barrier needed
#pragma unroll
    for (int nt = 0; nt < 4; nt++)
#pragma unroll
      for (int r = 0; r < 8; r++)
        Pt[wave][(r + 8 * (lane >> 4)) * 64 + nt * 16 + (lane & 15)] = (__bf16)s[nt][r];

    v16bf ap0 = load_a_frag(Pt[wave] + 0,  64, lane);
    v16bf ap1 = load_a_frag(Pt[wave] + 32, 64, lane);
#pragma unroll
    for (int dt = 0; dt < 4; dt++) {
      v16bf bv0 = load_b_frag(Vb + dt * 16 * 64 + 0,  64, lane);
      v16bf bv1 = load_b_frag(Vb + dt * 16 * 64 + 32, 64, lane);
      acc[dt] = wmma_bf16(ap0, bv0, acc[dt]);
      acc[dt] = wmma_bf16(ap1, bv1, acc[dt]);
    }
    __syncthreads();
  }

  // normalize and store ctx[b][l][h*64+d] (row-major [4096,1024] for the output GEMM)
  int b = bh >> 4, h = bh & 15;
#pragma unroll
  for (int dt = 0; dt < 4; dt++)
#pragma unroll
    for (int r = 0; r < 8; r++) {
      int row = qbase + r + 8 * (lane >> 4);
      int d = dt * 16 + (lane & 15);
      float o = acc[dt][r] / lrun[r];
      ctx[((size_t)(b * SEQ + row) * NH + h) * HD + d] = (__bf16)o;
    }
}

// ---------------- output projection: out(f32) = ctx(bf16) @ Wo (WoT given transposed) ----------------
__global__ __launch_bounds__(128) void mha_out_proj_kernel(const __bf16* __restrict__ X,
                                                           const __bf16* __restrict__ Wt,
                                                           float* __restrict__ out) {
  __shared__ __align__(32) __bf16 At[2][64 * 32];
  __shared__ __align__(32) __bf16 Bt[2][64 * 32];
  int tid = threadIdx.x, lane = tid & 31, wave = tid >> 5;
  int row0 = blockIdx.y * 64, col0 = blockIdx.x * 64;
  const int NK = D_MODEL / 32;
  const __bf16* Arow = X  + (size_t)row0 * D_MODEL;
  const __bf16* Brow = Wt + (size_t)col0 * D_MODEL;
  v8f c[2][2];
#pragma unroll
  for (int i = 0; i < 2; i++)
#pragma unroll
    for (int j = 0; j < 2; j++)
#pragma unroll
      for (int r = 0; r < 8; r++) c[i][j][r] = 0.f;

  int wrow = (wave >> 1) * 32, wcol = (wave & 1) * 32;

  if (wave == 0) {
    tdm_load_2d(At[0], Arow, D_MODEL, 64, 32, 64, D_MODEL);
    tdm_load_2d(Bt[0], Brow, D_MODEL, 64, 32, 64, D_MODEL);
  }

  for (int ks = 0; ks < NK; ks += 2) {
    if (wave == 0) {
      int kn = (ks + 1) * 32;
      tdm_load_2d(At[1], Arow + kn, D_MODEL, 64, 32, 64, D_MODEL);
      tdm_load_2d(Bt[1], Brow + kn, D_MODEL, 64, 32, 64, D_MODEL);
      __builtin_amdgcn_s_wait_tensorcnt(2);
    }
    __syncthreads();
    gemm_step(At[0], Bt[0], wrow, wcol, lane, c);
    __syncthreads();

    if (wave == 0) {
      if (ks + 2 < NK) {
        int kn = (ks + 2) * 32;
        tdm_load_2d(At[0], Arow + kn, D_MODEL, 64, 32, 64, D_MODEL);
        tdm_load_2d(Bt[0], Brow + kn, D_MODEL, 64, 32, 64, D_MODEL);
        __builtin_amdgcn_s_wait_tensorcnt(2);
      } else {
        __builtin_amdgcn_s_wait_tensorcnt(0);
      }
    }
    __syncthreads();
    gemm_step(At[1], Bt[1], wrow, wcol, lane, c);
    __syncthreads();
  }

#pragma unroll
  for (int mi = 0; mi < 2; mi++)
#pragma unroll
    for (int ni = 0; ni < 2; ni++)
#pragma unroll
      for (int r = 0; r < 8; r++) {
        int row = row0 + wrow + mi * 16 + r + 8 * (lane >> 4);
        int col = col0 + wcol + ni * 16 + (lane & 15);
        out[(size_t)row * D_MODEL + col] = c[mi][ni][r];
      }
}

extern "C" void kernel_launch(void* const* d_in, const int* in_sizes, int n_in,
                              void* d_out, int out_size, void* d_ws, size_t ws_size,
                              hipStream_t stream) {
  const float* x  = (const float*)d_in[0];
  const float* Wq = (const float*)d_in[1];
  const float* Wk = (const float*)d_in[2];
  const float* Wv = (const float*)d_in[3];
  const float* Wo = (const float*)d_in[4];

  const size_t NX = (size_t)MROWS * D_MODEL;    // 4 Mi elems
  const size_t NW = (size_t)D_MODEL * D_MODEL;  // 1 Mi elems

  __bf16* xb   = (__bf16*)d_ws;
  __bf16* wqT  = xb  + NX;
  __bf16* wkT  = wqT + NW;
  __bf16* wvT  = wkT + NW;
  __bf16* woT  = wvT + NW;
  __bf16* qb   = woT + NW;
  __bf16* kb   = qb  + NX;
  __bf16* vT   = kb  + NX;
  __bf16* ctx  = vT  + NX;   // total 24 Mi bf16 = 48 MiB

  mha_cvt_kernel<<<(int)(NX / 256), 256, 0, stream>>>(x, xb, (int)NX);
  dim3 tg(D_MODEL / 32, D_MODEL / 32), tb(32, 8);
  mha_cvtT_kernel<<<tg, tb, 0, stream>>>(Wq, wqT);
  mha_cvtT_kernel<<<tg, tb, 0, stream>>>(Wk, wkT);
  mha_cvtT_kernel<<<tg, tb, 0, stream>>>(Wv, wvT);
  mha_cvtT_kernel<<<tg, tb, 0, stream>>>(Wo, woT);

  dim3 gg(D_MODEL / 64, MROWS / 64);
  const float scale = 0.125f;  // hd^-0.5 = 1/8, folded into Q
  mha_proj_qkv_kernel<<<gg, 128, 0, stream>>>(xb, wqT, qb, scale, 0);
  mha_proj_qkv_kernel<<<gg, 128, 0, stream>>>(xb, wkT, kb, 1.0f, 0);
  mha_proj_qkv_kernel<<<gg, 128, 0, stream>>>(xb, wvT, vT, 1.0f, 1);

  mha_attn_kernel<<<dim3(SEQ / 64, NB * NH), 128, 0, stream>>>(qb, kb, vT, ctx);

  mha_out_proj_kernel<<<gg, 128, 0, stream>>>(ctx, woT, (float*)d_out);
}
